// PointEncoder_15075335209699
// MI455X (gfx1250) — compile-verified, matrix operand order checked by reading, not run on hardware
//
#include <hip/hip_runtime.h>
#include <cstdint>

// ---------------------------------------------------------------------------
// CDNA5 (gfx1250) point-encoder pipeline.
// All 1x1 convs lowered to bf16 WMMA GEMMs (v_wmma_f32_16x16x32_bf16),
// f32 accumulate. BatchNorm uses batch stats -> separate reduce + finalize;
// conv bias is absorbed by BN mean subtraction and skipped.
// Weights are pre-converted to zero-padded bf16 so the GEMM K-loop has NO
// guards: unconditional b128 global->LDS staging, plus speculative
// global_prefetch of the next K-step.
// ---------------------------------------------------------------------------

typedef __attribute__((ext_vector_type(16))) __bf16 v16bf;
typedef __attribute__((ext_vector_type(8)))  float  v8f;

static constexpr int  BB  = 8;
static constexpr int  NN  = 8192;
static constexpr int  KNB = 16;
static constexpr long MPT = (long)BB * NN;   // 65536 points
static constexpr long MKT = MPT * KNB;       // 1048576 edge samples
static constexpr float BN_EPS = 1e-5f;

// ---------------------------------------------------------------------------
// Weight pre-conversion: f32 [Co][Ci] -> bf16 [CoPad][CiPad], zero padded.
// ---------------------------------------------------------------------------
__global__ void wconv_kernel(const float* __restrict__ W, int Co, int Ci,
                             int CiPad, long total, __bf16* __restrict__ out)
{
    long idx = (long)blockIdx.x * blockDim.x + threadIdx.x;
    if (idx >= total) return;
    int co = (int)(idx / CiPad);
    int k  = (int)(idx % CiPad);
    out[idx] = (co < Co && k < Ci) ? (__bf16)W[(size_t)co * Ci + k] : (__bf16)0.0f;
}

// ---------------------------------------------------------------------------
// GEMM: Y[m][co] = sum_ci Wb[co][ci] * X[m][ci]
// X: bf16 [Mtot][ldx] (pad cols pre-zeroed), Wb: bf16 [CoPad][CiPad] (padded),
// Y: f32 [Mtot][Co].
// Tile: 256 points x 64 channels, K-step 32. 8 waves = 4 co-subtiles x
// 2 m-groups; each wave: 1 A fragment reused across 8 WMMAs per K-step.
// Fragment layout (16-bit operands, wave32): lane L -> row/col = L%16,
// hi = L>>4; v16bf elements 0..7 = K[hi*8..+7], 8..15 = K[16+hi*8..+7].
// ---------------------------------------------------------------------------
__global__ __launch_bounds__(256)
void gemm_wmma(const __bf16* __restrict__ X, int ldx,
               const __bf16* __restrict__ Wb, int Co, int CiPad,
               float* __restrict__ Y)
{
    __shared__ __bf16 sX[256][32];   // 16 KB point-major B tile
    __shared__ __bf16 sW[64][32];    //  4 KB weight A tile

    const int tid   = threadIdx.x;
    const int lane  = tid & 31;
    const int wid   = tid >> 5;
    const int lo16  = lane & 15;
    const int hi    = lane >> 4;
    const int cosub = wid & 3;       // 4 co subtiles of 16
    const int msub  = wid >> 2;      // 2 m groups of 128
    const long m0   = (long)blockIdx.x * 256;
    const int  co0  = blockIdx.y * 64;

    v8f acc[8];
    #pragma unroll
    for (int f = 0; f < 8; ++f) acc[f] = v8f{};

    const int xrow = tid >> 2, xq = tid & 3;     // X tile: 4 uint4/thread
    const int wrow = tid >> 2, wq = tid & 3;     // W tile: 1 uint4/thread
    const __bf16* xbase = X + (m0 + xrow) * (long)ldx + xq * 8;
    const __bf16* wbase = Wb + (size_t)(co0 + wrow) * CiPad + wq * 8;
    const long xstep = 64 * (long)ldx;           // 64 rows per staging slice

    for (int k0 = 0; k0 < CiPad; k0 += 32) {
        __syncthreads();                          // previous fragments consumed
        #pragma unroll
        for (int i = 0; i < 4; ++i) {
            uint4 v = *reinterpret_cast<const uint4*>(xbase + i * xstep + k0);
            *reinterpret_cast<uint4*>(&sX[xrow + i * 64][xq * 8]) = v;
            // speculative prefetch of next K-step (drops silently if OOB)
            __builtin_prefetch(xbase + i * xstep + k0 + 32, 0, 1);
        }
        *reinterpret_cast<uint4*>(&sW[wrow][wq * 8]) =
            *reinterpret_cast<const uint4*>(wbase + k0);
        __syncthreads();

        union FragU { v16bf v; uint4 q[2]; };
        FragU af;
        const int ar = cosub * 16 + lo16;
        af.q[0] = *reinterpret_cast<const uint4*>(&sW[ar][hi * 8]);
        af.q[1] = *reinterpret_cast<const uint4*>(&sW[ar][16 + hi * 8]);

        #pragma unroll
        for (int f = 0; f < 8; ++f) {
            FragU bfr;
            const int br = msub * 128 + f * 16 + lo16;
            bfr.q[0] = *reinterpret_cast<const uint4*>(&sX[br][hi * 8]);
            bfr.q[1] = *reinterpret_cast<const uint4*>(&sX[br][16 + hi * 8]);
            acc[f] = __builtin_amdgcn_wmma_f32_16x16x32_bf16(
                false, af.v, false, bfr.v, (short)0, acc[f], false, false);
        }
    }

    // epilogue: D element r of lane -> co = co0 + cosub*16 + hi*8 + r
    const int coBase = co0 + cosub * 16 + hi * 8;
    if (coBase < Co) {                            // Co always multiple of 8
        #pragma unroll
        for (int f = 0; f < 8; ++f) {
            long m = m0 + msub * 128 + f * 16 + lo16;
            float* dst = Y + m * (long)Co + coBase;
            float4 a0, a1;
            a0.x = acc[f][0]; a0.y = acc[f][1]; a0.z = acc[f][2]; a0.w = acc[f][3];
            a1.x = acc[f][4]; a1.y = acc[f][5]; a1.z = acc[f][6]; a1.w = acc[f][7];
            *reinterpret_cast<float4*>(dst)     = a0;
            *reinterpret_cast<float4*>(dst + 4) = a1;
        }
    }
}

// ---------------------------------------------------------------------------
// Edge feature build: e[m*K+k] = [x_center (C), x_nbr - x_center (C)] bf16,
// pad columns [2C, lde) zero-filled so the GEMM needs no guards.
// ---------------------------------------------------------------------------
__global__ void gather_edge(const float* __restrict__ x, int C,
                            const int* __restrict__ knn,
                            __bf16* __restrict__ e, int lde)
{
    long mk = (long)blockIdx.x * blockDim.x + threadIdx.x;
    if (mk >= MKT) return;
    long m = mk / KNB;
    long b = m / NN;
    int  idx = knn[mk];
    const float* cen = x + m * (long)C;
    const float* nbr = x + (b * NN + idx) * (long)C;
    __bf16* dst = e + mk * (long)lde;
    for (int c = 0; c < C; ++c) {
        float cv = cen[c], nv = nbr[c];
        dst[c]     = (__bf16)cv;
        dst[C + c] = (__bf16)(nv - cv);
    }
    for (int c = 2 * C; c < lde; ++c) dst[c] = (__bf16)0.0f;
}

// ---------------------------------------------------------------------------
// Per-channel sum / sumsq over all rows (batch-stat BN).
// ---------------------------------------------------------------------------
__global__ void zero_kernel(float* __restrict__ p, int n)
{
    int i = blockIdx.x * blockDim.x + threadIdx.x;
    if (i < n) p[i] = 0.0f;
}

__global__ void stats_kernel(const float* __restrict__ Y, long Mtot, int Co,
                             float* __restrict__ stats)
{
    int co = blockIdx.x * 256 + threadIdx.x;
    if (co >= Co) return;
    long chunk  = Mtot / gridDim.y;
    long mstart = (long)blockIdx.y * chunk;
    float s = 0.0f, s2 = 0.0f;
    const float* p = Y + mstart * (long)Co + co;
    for (long i = 0; i < chunk; ++i) {
        float v = p[i * Co];
        s += v; s2 += v * v;
    }
    atomicAdd(&stats[co], s);
    atomicAdd(&stats[Co + co], s2);
}

__global__ void bnfin_kernel(const float* __restrict__ stats,
                             const float* __restrict__ gam,
                             const float* __restrict__ bet,
                             long Mtot, int Co, float* __restrict__ ss)
{
    int co = blockIdx.x * blockDim.x + threadIdx.x;
    if (co >= Co) return;
    float inv  = 1.0f / (float)Mtot;
    float mean = stats[co] * inv;
    float var  = stats[Co + co] * inv - mean * mean;
    float sc   = gam[co] * rsqrtf(var + BN_EPS);
    ss[co]      = sc;
    ss[Co + co] = bet[co] - mean * sc;
}

// BN+ReLU, then max over K neighbors (edge conv). Writes next-layer x (f32)
// and the dense-concat encode output (bf16 column slice).
__global__ void bn_relu_max_edge(const float* __restrict__ Y,
                                 const float* __restrict__ ss,
                                 float* __restrict__ xout,
                                 __bf16* __restrict__ cat, int catOff)
{
    long idx = (long)blockIdx.x * blockDim.x + threadIdx.x;
    if (idx >= MPT * 24) return;
    long m = idx / 24;
    int  co = (int)(idx % 24);
    float sc = ss[co], sh = ss[24 + co];
    const float* src = Y + (m * KNB) * 24 + co;
    float best = 0.0f;                       // relu outputs are >= 0
    #pragma unroll
    for (int k = 0; k < KNB; ++k)
        best = fmaxf(best, fmaxf(src[(long)k * 24] * sc + sh, 0.0f));
    xout[idx] = best;
    cat[m * 96 + catOff + co] = (__bf16)best;
}

// BN+ReLU -> bf16 column slice of next GEMM input.
__global__ void bn_relu_store(const float* __restrict__ Y, long Mtot, int Co,
                              const float* __restrict__ ss,
                              __bf16* __restrict__ out, int ldo, int colOff)
{
    long idx = (long)blockIdx.x * blockDim.x + threadIdx.x;
    if (idx >= Mtot * Co) return;
    long m = idx / Co;
    int  co = (int)(idx % Co);
    float v = fmaxf(Y[idx] * ss[co] + ss[Co + co], 0.0f);
    out[m * (long)ldo + colOff + co] = (__bf16)v;
}

// Exact top-2 / top-1 pool over N per (batch, channel); BN+ReLU on the fly.
// blockDim.x == Co, gridDim.x == B.
__global__ void topk_pool(const float* __restrict__ Y, int Co,
                          const float* __restrict__ ss, int topk,
                          float* __restrict__ gout)
{
    int b  = blockIdx.x;
    int co = threadIdx.x;
    float sc = ss[co], sh = ss[Co + co];
    float b1 = -1e30f, b2 = -1e30f;
    const float* src = Y + ((long)b * NN) * Co + co;
    for (int n = 0; n < NN; ++n) {
        float v = fmaxf(src[(long)n * Co] * sc + sh, 0.0f);
        if (v > b1) { b2 = b1; b1 = v; }
        else if (v > b2) { b2 = v; }
    }
    gout[b * Co + co] = (topk == 2) ? 0.5f * (b1 + b2) : b1;
}

// Tiny FC (B=8 rows) + BN over batch + ReLU. grid = Co, block = 64.
__global__ __launch_bounds__(64)
void fc_bn_relu(const float* __restrict__ gin, const float* __restrict__ W,
                const float* __restrict__ gam, const float* __restrict__ bet,
                int Ci, int Co, float* __restrict__ gout)
{
    __shared__ float red[8][64];
    int co = blockIdx.x, t = threadIdx.x;
    float a[8];
    #pragma unroll
    for (int b = 0; b < 8; ++b) a[b] = 0.0f;
    for (int ci = t; ci < Ci; ci += 64) {
        float w = W[(size_t)co * Ci + ci];
        #pragma unroll
        for (int b = 0; b < 8; ++b) a[b] += gin[b * Ci + ci] * w;
    }
    #pragma unroll
    for (int b = 0; b < 8; ++b) red[b][t] = a[b];
    __syncthreads();
    if (t == 0) {
        float y[8]; float mean = 0.0f;
        #pragma unroll
        for (int b = 0; b < 8; ++b) {
            float s = 0.0f;
            for (int i = 0; i < 64; ++i) s += red[b][i];
            y[b] = s; mean += s;
        }
        mean *= 0.125f;
        float var = 0.0f;
        #pragma unroll
        for (int b = 0; b < 8; ++b) { float d = y[b] - mean; var += d * d; }
        var *= 0.125f;
        float sc = gam[co] * rsqrtf(var + BN_EPS);
        float sh = bet[co];
        #pragma unroll
        for (int b = 0; b < 8; ++b)
            gout[b * Co + co] = fmaxf((y[b] - mean) * sc + sh, 0.0f);
    }
}

// Broadcast a per-batch global code into a column slice of a bf16 GEMM input.
__global__ void bcast_kernel(const float* __restrict__ g, int Cg,
                             __bf16* __restrict__ out, int ldo, int colOff)
{
    long idx = (long)blockIdx.x * blockDim.x + threadIdx.x;
    if (idx >= MPT * (long)Cg) return;
    long m = idx / Cg;
    int  c = (int)(idx % Cg);
    long b = m / NN;
    out[m * (long)ldo + colOff + c] = (__bf16)g[b * Cg + c];
}

// Final BN+ReLU, written in the reference's (B, 128, N) layout as f32.
__global__ void bn_relu_out(const float* __restrict__ Y,
                            const float* __restrict__ ss,
                            float* __restrict__ out)
{
    long idx = (long)blockIdx.x * blockDim.x + threadIdx.x;
    if (idx >= MPT * 128) return;
    long m = idx / 128;
    int  co = (int)(idx % 128);
    long b = m / NN, n = m % NN;
    float v = fmaxf(Y[idx] * ss[co] + ss[128 + co], 0.0f);
    out[((long)b * 128 + co) * NN + n] = v;
}

// ---------------------------------------------------------------------------
// Host orchestration.
// Param flattening (jax pytree, sorted dict keys; pconv order W,b,be,g):
//   [0]=pos [1]=knn, conv_1@2, conv_2@6, conv_3@10, encode[l]@14+4l,
//   shift_1: conv_a@30 conv_b@34 fc@38, shift_2: @42/@46/@50, shift_3: @54/@58/@62
// ---------------------------------------------------------------------------
extern "C" void kernel_launch(void* const* d_in, const int* in_sizes, int n_in,
                              void* d_out, int out_size, void* d_ws, size_t ws_size,
                              hipStream_t stream)
{
    (void)in_sizes; (void)n_in; (void)out_size; (void)ws_size;
    const float* pos = (const float*)d_in[0];
    const int*   knn = (const int*)d_in[1];
    auto P = [&](int i) { return (const float*)d_in[i]; };

    char* ws = (char*)d_ws;
    size_t off = 0;
    auto alloc = [&](size_t bytes) -> void* {
        void* p = ws + off;
        off += (bytes + 255) & ~size_t(255);
        return p;
    };
    // bufE doubles as edge-feature buffer (MKT x 64 bf16) and conv_b input (MPT x 1280 bf16)
    __bf16* bufE   = (__bf16*)alloc(sizeof(__bf16) * (size_t)MPT * 1280);
    // bufY doubles as edge GEMM output (MKT x 24 f32) and big GEMM output (MPT x 512 f32)
    float*  bufY   = (float*)alloc(sizeof(float) * (size_t)MPT * 512);
    __bf16* bufAct = (__bf16*)alloc(sizeof(__bf16) * (size_t)MPT * 512);
    __bf16* bufCat = (__bf16*)alloc(sizeof(__bf16) * (size_t)MPT * 96);
    float*  bufX   = (float*)alloc(sizeof(float) * (size_t)MPT * 24);
    __bf16* bufWb  = (__bf16*)alloc(sizeof(__bf16) * (size_t)512 * 1280);
    float*  stats  = (float*)alloc(sizeof(float) * 1024);
    float*  ssbuf  = (float*)alloc(sizeof(float) * 1024);
    float*  gpre   = (float*)alloc(sizeof(float) * 8 * 512);
    float*  gfc1   = (float*)alloc(sizeof(float) * 8 * 256);
    float*  gfc2   = (float*)alloc(sizeof(float) * 8 * 512);
    float*  gfc3   = (float*)alloc(sizeof(float) * 8 * 512);

    auto gemm = [&](const __bf16* X, int ldx, const float* W, int Co, int Ci,
                    long Mtot, float* Y) {
        int CiPad = (Ci + 31) & ~31;
        int CoPad = (Co + 63) & ~63;
        long nw = (long)CoPad * CiPad;
        wconv_kernel<<<(unsigned)((nw + 255) / 256), 256, 0, stream>>>(
            W, Co, Ci, CiPad, nw, bufWb);
        dim3 g((unsigned)(Mtot / 256), (unsigned)(CoPad / 64));
        gemm_wmma<<<g, 256, 0, stream>>>(X, ldx, bufWb, Co, CiPad, Y);
    };
    auto stats_bn = [&](const float* Y, long Mtot, int Co,
                        const float* gam, const float* bet) {
        zero_kernel<<<(2 * Co + 255) / 256, 256, 0, stream>>>(stats, 2 * Co);
        dim3 g((unsigned)((Co + 255) / 256), 64u);
        stats_kernel<<<g, 256, 0, stream>>>(Y, Mtot, Co, stats);
        bnfin_kernel<<<(Co + 255) / 256, 256, 0, stream>>>(stats, gam, bet, Mtot, Co, ssbuf);
    };
    auto store = [&](const float* Y, long Mtot, int Co, __bf16* out, int ldo, int colOff) {
        long n = Mtot * Co;
        bn_relu_store<<<(unsigned)((n + 255) / 256), 256, 0, stream>>>(
            Y, Mtot, Co, ssbuf, out, ldo, colOff);
    };
    auto bcast = [&](const float* g, int Cg, __bf16* out, int ldo, int colOff) {
        long n = MPT * (long)Cg;
        bcast_kernel<<<(unsigned)((n + 255) / 256), 256, 0, stream>>>(g, Cg, out, ldo, colOff);
    };

    // ---------------- EncodeNet: 4 edge convs ----------------
    const float* xcur = pos;
    int C = 3;
    for (int l = 0; l < 4; ++l) {
        int pw = 14 + 4 * l;
        int Ce = 2 * C;
        gather_edge<<<(unsigned)((MKT + 255) / 256), 256, 0, stream>>>(xcur, C, knn, bufE, 64);
        gemm(bufE, 64, P(pw), 24, Ce, MKT, bufY);
        stats_bn(bufY, MKT, 24, P(pw + 3), P(pw + 2));
        bn_relu_max_edge<<<(unsigned)((MPT * 24 + 255) / 256), 256, 0, stream>>>(
            bufY, ssbuf, bufX, bufCat, l * 24);
        xcur = bufX; C = 24;
    }

    // ---------------- conv_1: 96 -> 128 ----------------
    gemm(bufCat, 96, P(2), 128, 96, MPT, bufY);
    stats_bn(bufY, MPT, 128, P(5), P(4));
    store(bufY, MPT, 128, bufAct, 128, 0);

    // ---------------- shift_1: 128 -> 256, topk=2 ----------------
    gemm(bufAct, 128, P(30), 256, 128, MPT, bufY);
    stats_bn(bufY, MPT, 256, P(33), P(32));
    store(bufY, MPT, 256, bufE, 512, 0);
    topk_pool<<<8, 256, 0, stream>>>(bufY, 256, ssbuf, 2, gpre);
    fc_bn_relu<<<256, 64, 0, stream>>>(gpre, P(38), P(41), P(40), 256, 256, gfc1);
    bcast(gfc1, 256, bufE, 512, 256);
    gemm(bufE, 512, P(34), 256, 512, MPT, bufY);
    stats_bn(bufY, MPT, 256, P(37), P(36));
    store(bufY, MPT, 256, bufAct, 256, 0);

    // ---------------- shift_2: 256 -> 512, topk=1, + g1 ----------------
    gemm(bufAct, 256, P(42), 512, 256, MPT, bufY);
    stats_bn(bufY, MPT, 512, P(45), P(44));
    store(bufY, MPT, 512, bufE, 1280, 0);
    topk_pool<<<8, 512, 0, stream>>>(bufY, 512, ssbuf, 1, gpre);
    fc_bn_relu<<<512, 64, 0, stream>>>(gpre, P(50), P(53), P(52), 512, 512, gfc2);
    bcast(gfc2, 512, bufE, 1280, 512);
    bcast(gfc1, 256, bufE, 1280, 1024);
    gemm(bufE, 1280, P(46), 512, 1280, MPT, bufY);
    stats_bn(bufY, MPT, 512, P(49), P(48));
    store(bufY, MPT, 512, bufAct, 512, 0);

    // ---------------- shift_3: 512 -> 512, topk=1 ----------------
    gemm(bufAct, 512, P(54), 512, 512, MPT, bufY);
    stats_bn(bufY, MPT, 512, P(57), P(56));
    store(bufY, MPT, 512, bufE, 1024, 0);
    topk_pool<<<8, 512, 0, stream>>>(bufY, 512, ssbuf, 1, gpre);
    fc_bn_relu<<<512, 64, 0, stream>>>(gpre, P(62), P(65), P(64), 512, 512, gfc3);
    bcast(gfc3, 512, bufE, 1024, 512);
    gemm(bufE, 1024, P(58), 512, 1024, MPT, bufY);
    stats_bn(bufY, MPT, 512, P(61), P(60));
    store(bufY, MPT, 512, bufAct, 512, 0);

    // ---------------- conv_2: 512 -> 256 ----------------
    gemm(bufAct, 512, P(6), 256, 512, MPT, bufY);
    stats_bn(bufY, MPT, 256, P(9), P(8));
    store(bufY, MPT, 256, bufAct, 256, 0);

    // ---------------- conv_3: 256 -> 128, final output ----------------
    gemm(bufAct, 256, P(10), 128, 256, MPT, bufY);
    stats_bn(bufY, MPT, 128, P(13), P(12));
    bn_relu_out<<<(unsigned)((MPT * 128 + 255) / 256), 256, 0, stream>>>(
        bufY, ssbuf, (float*)d_out);
}
